// MultiheadAttention_70076686402241
// MI455X (gfx1250) — matrix operand order repaired and checked by reading, not compiled, not run on hardware
//
#include <hip/hip_runtime.h>
#include <math.h>

typedef __attribute__((ext_vector_type(16))) _Float16 v16h;
typedef __attribute__((ext_vector_type(8)))  _Float16 v8h;
typedef __attribute__((ext_vector_type(4)))  _Float16 v4h;
typedef __attribute__((ext_vector_type(8)))  float    v8f;

#define H_   16
#define S_   4096
#define DH_  128
#define DM_  (H_ * DH_)
#define BM   128            // query rows per block (8 waves x 16 rows)
#define BN   32             // keys per kv tile
#define VSTR 34             // padded stride (f16 units) for transposed V in LDS
#define VPAD(d) ((((d) >> 6) & 1) << 1)   // +2 f16 for d>=64 (uniform in g for frag reads)
#define SCALE 0.08838834764831843f        // 1/sqrt(128)

__device__ __forceinline__ v4h cvt4(float4 x) {
    return (v4h){ (_Float16)x.x, (_Float16)x.y, (_Float16)x.z, (_Float16)x.w };
}

__launch_bounds__(256, 1)
__global__ void fa_fwd_70076686402241(const float* __restrict__ Q,
                                      const float* __restrict__ K,
                                      const float* __restrict__ V,
                                      float* __restrict__ O)
{
    // double-buffered f16 tiles
    __shared__ __attribute__((aligned(32))) _Float16 sKh[2][BN * DH_];        // row-major [k][d]
    __shared__ __attribute__((aligned(16))) _Float16 sVt[2][DH_ * VSTR + 2];  // transposed [d][k]
    __shared__ __attribute__((aligned(16))) _Float16 sP[8][16 * BN];          // per-wave P scratch

    const int tid  = threadIdx.x;
    const int wid  = tid >> 5;
    const int lane = tid & 31;
    const int r    = lane & 15;    // A-row / B,C,D column
    const int hh   = lane >> 4;    // lane half

    const int qtile = blockIdx.x;
    const int head  = blockIdx.y;

    const int wqb = qtile * BM + wid * 16;     // wave's first global q row

    // ---- load Q fragments (A 16x32 layout), fp32 -> f16, SCALE folded in ----
    const float* qptr = Q + ((size_t)head * S_ + (wqb + r)) * DH_;
    v16h Qa[4];
#pragma unroll
    for (int f = 0; f < 4; ++f) {
#pragma unroll
        for (int e = 0; e < 8; ++e) {
            Qa[f][e]     = (_Float16)(qptr[32 * f + 16 * hh + e] * SCALE);
            Qa[f][8 + e] = (_Float16)(qptr[32 * f + 16 * hh + 8 + e] * SCALE);
        }
    }

    // constant all-ones B matrix: rowsum(P) = P x ones via the matrix pipe
    v16h OnesB;
#pragma unroll
    for (int e = 0; e < 16; ++e) OnesB[e] = (_Float16)1.0f;

    v8f Of[8];
#pragma unroll
    for (int g = 0; g < 8; ++g) Of[g] = (v8f){};
    float rmax[8], rsum[8];
#pragma unroll
    for (int vv = 0; vv < 8; ++vv) { rmax[vv] = -INFINITY; rsum[vv] = 0.0f; }

    const int    ntiles = (qtile * BM + BM) / BN;   // causal limit
    const float* kbase0 = K + (size_t)head * S_ * DH_;
    const float* vbase0 = V + (size_t)head * S_ * DH_;

    // register prefetch of tile 0: 2 tasks/thread, each = row pair x float4 col
    float4 kA[2], kB[2], vA[2], vB[2];
#pragma unroll
    for (int i = 0; i < 2; ++i) {
        const int task = tid + 256 * i;
        const int dq = task & 31, a = task >> 5;
        kA[i] = ((const float4*)kbase0)[(2 * a) * 32 + dq];
        kB[i] = ((const float4*)kbase0)[(2 * a + 1) * 32 + dq];
        vA[i] = ((const float4*)vbase0)[(2 * a) * 32 + dq];
        vB[i] = ((const float4*)vbase0)[(2 * a + 1) * 32 + dq];
    }

    for (int t = 0; t < ntiles; ++t) {
        const int b = t & 1;

        // ---- convert prefetched registers -> f16 LDS ----
#pragma unroll
        for (int i = 0; i < 2; ++i) {
            const int task = tid + 256 * i;
            const int dq = task & 31, a = task >> 5;
            *(v4h*)&sKh[b][(2 * a) * DH_ + 4 * dq]     = cvt4(kA[i]);
            *(v4h*)&sKh[b][(2 * a + 1) * DH_ + 4 * dq] = cvt4(kB[i]);
#pragma unroll
            for (int e = 0; e < 4; ++e) {
                const int d = 4 * dq + e;
                union { _Float16 h[2]; unsigned int u; } pk;
                pk.h[0] = (_Float16)((const float*)&vA[i])[e];   // V[2a  ][d]
                pk.h[1] = (_Float16)((const float*)&vB[i])[e];   // V[2a+1][d]
                *(unsigned int*)&sVt[b][d * VSTR + VPAD(d) + 2 * a] = pk.u;
            }
        }
        __syncthreads();

        // ---- issue global loads for next tile (latency hidden behind compute) ----
        if (t + 1 < ntiles) {
            const float4* ks = (const float4*)(kbase0 + (size_t)(t + 1) * BN * DH_);
            const float4* vs = (const float4*)(vbase0 + (size_t)(t + 1) * BN * DH_);
#pragma unroll
            for (int i = 0; i < 2; ++i) {
                const int task = tid + 256 * i;
                const int dq = task & 31, a = task >> 5;
                kA[i] = ks[(2 * a) * 32 + dq];
                kB[i] = ks[(2 * a + 1) * 32 + dq];
                vA[i] = vs[(2 * a) * 32 + dq];
                vB[i] = vs[(2 * a + 1) * 32 + dq];
            }
        }

        const int kv0 = t * BN;
        if (kv0 <= wqb + 15) {   // wave-uniform: tile not fully above the diagonal
            // ---- S = Q . K^T (two 16-key groups); B-frag = 32 contiguous bytes ----
            v8f acc[2];
#pragma unroll
            for (int ng = 0; ng < 2; ++ng) {
                v8f a = (v8f){};
#pragma unroll
                for (int f = 0; f < 4; ++f) {
                    v16h Kb = *(const v16h*)&sKh[b][(ng * 16 + r) * DH_ + 32 * f + 16 * hh];
                    a = __builtin_amdgcn_wmma_f32_16x16x32_f16(
                            false, Qa[f], false, Kb, (short)0, a, false, false);
                }
                acc[ng] = a;
            }

            // ---- causal mask: only the single straddling tile needs it ----
            if (kv0 + BN - 1 > wqb) {
#pragma unroll
                for (int ng = 0; ng < 2; ++ng) {
                    const int kg = kv0 + ng * 16 + r;
#pragma unroll
                    for (int vv = 0; vv < 8; ++vv) {
                        const int mrow = wqb + vv + 8 * hh;
                        acc[ng][vv] = (kg > mrow) ? -INFINITY : acc[ng][vv];
                    }
                }
            }

            // ---- row max: level-batched 16-lane tree (8 independent bpermutes/level) ----
            float tmx[8];
#pragma unroll
            for (int vv = 0; vv < 8; ++vv) tmx[vv] = fmaxf(acc[0][vv], acc[1][vv]);
#pragma unroll
            for (int off = 1; off < 16; off <<= 1) {
                float sh[8];
#pragma unroll
                for (int vv = 0; vv < 8; ++vv) sh[vv] = __shfl_xor(tmx[vv], off, 32);
#pragma unroll
                for (int vv = 0; vv < 8; ++vv) tmx[vv] = fmaxf(tmx[vv], sh[vv]);
            }

            // ---- batched exponentials (TRANS-pipe ILP), then stores + O rescale ----
            float corr[8], p0a[8], p1a[8];
#pragma unroll
            for (int vv = 0; vv < 8; ++vv) {
                const float mnew = fmaxf(rmax[vv], tmx[vv]);
                corr[vv] = __expf(rmax[vv] - mnew);
                p0a[vv]  = __expf(acc[0][vv] - mnew);
                p1a[vv]  = __expf(acc[1][vv] - mnew);
                rmax[vv] = mnew;
            }
            _Float16* sp = &sP[wid][0];
#pragma unroll
            for (int vv = 0; vv < 8; ++vv) {
                sp[(vv + 8 * hh) * BN + r]      = (_Float16)p0a[vv];
                sp[(vv + 8 * hh) * BN + 16 + r] = (_Float16)p1a[vv];
            }
#pragma unroll
            for (int g = 0; g < 8; ++g)
#pragma unroll
                for (int vv = 0; vv < 8; ++vv) Of[g][vv] *= corr[vv];

            // ---- reload P in A 16x32 layout (same-wave DS ops in-order) ----
            v8h plo = *(const v8h*)&sp[r * BN + 16 * hh];          // 16-B aligned
            v8h phi = *(const v8h*)&sp[r * BN + 16 * hh + 8];
            v16h Pa = __builtin_shufflevector(plo, phi,
                        0, 1, 2, 3, 4, 5, 6, 7, 8, 9, 10, 11, 12, 13, 14, 15);

            // ---- row sums on the matrix pipe: P x ones (broadcast into C-layout) ----
            v8f psum = __builtin_amdgcn_wmma_f32_16x16x32_f16(
                           false, Pa, false, OnesB, (short)0, (v8f){}, false, false);
#pragma unroll
            for (int vv = 0; vv < 8; ++vv)
                rsum[vv] = rsum[vv] * corr[vv] + psum[vv];

            // ---- O += P . V : fragment bases as a pointer walk (1 add per g) ----
            const _Float16* vbase = &sVt[b][r * VSTR + 16 * hh];   // g=0 base (d0 = r)
#pragma unroll
            for (int g = 0; g < 8; ++g) {
                const unsigned int* vp = (const unsigned int*)(g >= 4 ? vbase + 2 : vbase);
                union { unsigned int u[8]; v16h h; } vb;
#pragma unroll
                for (int j = 0; j < 8; ++j) vb.u[j] = vp[j];
                Of[g] = __builtin_amdgcn_wmma_f32_16x16x32_f16(
                            false, Pa, false, vb.h, (short)0, Of[g], false, false);
                vbase += 16 * VSTR;
            }
        }
        __syncthreads();   // all waves done reading buf b before it is overwritten
    }

    // ---- epilogue: normalize and store (output is (S, H*DH)) ----
#pragma unroll
    for (int vv = 0; vv < 8; ++vv) {
        const float inv  = 1.0f / rsum[vv];
        const int   mrow = wqb + vv + 8 * hh;
        float* orow = O + (size_t)mrow * DM_ + head * DH_;
#pragma unroll
        for (int g = 0; g < 8; ++g)
            orow[16 * g + r] = Of[g][vv] * inv;
    }
}

extern "C" void kernel_launch(void* const* d_in, const int* in_sizes, int n_in,
                              void* d_out, int out_size, void* d_ws, size_t ws_size,
                              hipStream_t stream) {
    (void)in_sizes; (void)n_in; (void)d_ws; (void)ws_size; (void)out_size;
    const float* q = (const float*)d_in[0];
    const float* k = (const float*)d_in[1];
    const float* v = (const float*)d_in[2];
    float* o = (float*)d_out;
    dim3 grid(S_ / BM, H_, 1);
    fa_fwd_70076686402241<<<grid, dim3(256, 1, 1), 0, stream>>>(q, k, v, o);
}
